// EvolvingGNN_ODE_44933947851156
// MI455X (gfx1250) — matrix operand order repaired and checked by reading, not compiled, not run on hardware
//
#include <hip/hip_runtime.h>
#include <hip/hip_bf16.h>

typedef __attribute__((ext_vector_type(16))) __bf16 v16bf;
typedef __attribute__((ext_vector_type(8)))  float  v8f;

#define D_H   64
#define D_E   16
#define T_STEPS 8
#define FLAT  4096
#define BNECK 1024

__device__ __forceinline__ unsigned pack2bf(float lo, float hi) {
    union { __bf16 h[2]; unsigned u; } p;
    p.h[0] = (__bf16)lo;
    p.h[1] = (__bf16)hi;
    return p.u;
}

// put 8 floats (two float4 runs) into a[base..base+7]
__device__ __forceinline__ void put8(v16bf& a, int base, float4 x, float4 y) {
    a[base + 0] = (__bf16)x.x; a[base + 1] = (__bf16)x.y;
    a[base + 2] = (__bf16)x.z; a[base + 3] = (__bf16)x.w;
    a[base + 4] = (__bf16)y.x; a[base + 5] = (__bf16)y.y;
    a[base + 6] = (__bf16)y.z; a[base + 7] = (__bf16)y.w;
}

__device__ __forceinline__ float4 relu4(float4 v) {
    v.x = fmaxf(v.x, 0.0f); v.y = fmaxf(v.y, 0.0f);
    v.z = fmaxf(v.z, 0.0f); v.w = fmaxf(v.w, 0.0f);
    return v;
}

// ---------------------------------------------------------------------------
// Init: w_cur = init_w ; deg = 1 (self loop) ; h = 0
// ---------------------------------------------------------------------------
__global__ void init_state(const float* __restrict__ init_w,
                           float* __restrict__ w_cur,
                           float* __restrict__ deg,
                           float* __restrict__ h,
                           int N) {
    long long i = (long long)blockIdx.x * blockDim.x + threadIdx.x;
    long long tot = (long long)N * D_H;
    if (i < tot)  h[i] = 0.0f;
    if (i < N)    deg[i] = 1.0f;
    if (i < FLAT) w_cur[i] = init_w[i];
}

__global__ void deg_accum(const int* __restrict__ col, float* __restrict__ deg, int E) {
    int e = blockIdx.x * blockDim.x + threadIdx.x;
    if (e < E) atomicAdd(&deg[col[e]], 1.0f);
}

__global__ void deg_to_dis(float* __restrict__ deg, int N) {
    int i = blockIdx.x * blockDim.x + threadIdx.x;
    if (i < N) {
        float d = deg[i];
        deg[i] = d > 0.0f ? rsqrtf(d) : 0.0f;
    }
}

// ---------------------------------------------------------------------------
// ODE helpers
// ---------------------------------------------------------------------------
__global__ void ode_arg(const float* __restrict__ w,
                        const float* __restrict__ k1,
                        const float* __restrict__ k2,
                        const float* __restrict__ k3,
                        float c1, float c2, float c3,
                        float* __restrict__ arg) {
    int i = blockIdx.x * blockDim.x + threadIdx.x;
    if (i >= FLAT) return;
    float v = w[i];
    if (c1 != 0.0f) v += c1 * k1[i];
    if (c2 != 0.0f) v += c2 * k2[i];
    if (c3 != 0.0f) v += c3 * k3[i];
    arg[i] = v;
}

__global__ void ode_hidden(const float* __restrict__ arg,
                           const float* __restrict__ W1,
                           const float* __restrict__ b1,
                           float* __restrict__ hidden) {
    int wid  = (blockIdx.x * blockDim.x + threadIdx.x) >> 5;
    int lane = threadIdx.x & 31;
    if (wid >= BNECK) return;
    float acc = 0.0f;
    for (int i = lane; i < FLAT; i += 32)
        acc += arg[i] * W1[(long long)i * BNECK + wid];
    #pragma unroll
    for (int m = 16; m >= 1; m >>= 1) acc += __shfl_xor(acc, m, 32);
    if (lane == 0) hidden[wid] = tanhf(acc + b1[wid]);
}

__global__ void ode_out(const float* __restrict__ hidden,
                        const float* __restrict__ W2,
                        const float* __restrict__ b2,
                        float* __restrict__ kout) {
    int wid  = (blockIdx.x * blockDim.x + threadIdx.x) >> 5;
    int lane = threadIdx.x & 31;
    if (wid >= FLAT) return;
    float acc = 0.0f;
    for (int i = lane; i < BNECK; i += 32)
        acc += hidden[i] * W2[(long long)i * FLAT + wid];
    #pragma unroll
    for (int m = 16; m >= 1; m >>= 1) acc += __shfl_xor(acc, m, 32);
    if (lane == 0) kout[wid] = acc + b2[wid];
}

__global__ void ode_combine(float* __restrict__ w,
                            const float* __restrict__ k1,
                            const float* __restrict__ k2,
                            const float* __restrict__ k3,
                            const float* __restrict__ k4,
                            float hh) {
    int i = blockIdx.x * blockDim.x + threadIdx.x;
    if (i >= FLAT) return;
    w[i] += hh * (k1[i] + 3.0f * (k2[i] + k3[i]) + k4[i]) * 0.125f;
}

// ---------------------------------------------------------------------------
// xw = x_last @ W  (N x 64)(64 x 64), bf16 WMMA, wave per 16 rows.
// W staged in LDS pre-swizzled into WMMA B-fragment layout:
//   frag[(kc*4+nt)*32 + lane] = 16 bf16 = W[kc*32 + (lane>>4)*8 + (i>>3)*16 + (i&7)][nt*16 + (lane&15)]
// ---------------------------------------------------------------------------
__global__ void xw_wmma(const float* __restrict__ x_last,
                        const float* __restrict__ w_cur,
                        float* __restrict__ xw, int N) {
    __shared__ unsigned frag[2 * 4 * 32 * 8];   // 8 KB
    for (int idx = threadIdx.x; idx < 2 * 4 * 32 * 8; idx += blockDim.x) {
        int j    = idx & 7;
        int lane = (idx >> 3) & 31;
        int nt   = (idx >> 8) & 3;
        int kc   = idx >> 10;
        int n  = nt * 16 + (lane & 15);
        int kb = kc * 32 + (lane >> 4) * 8 + (j >> 2) * 16 + (j & 3) * 2;
        frag[idx] = pack2bf(w_cur[kb * D_H + n], w_cur[(kb + 1) * D_H + n]);
    }
    __syncthreads();

    int wid  = (blockIdx.x * blockDim.x + threadIdx.x) >> 5;
    int lane = threadIdx.x & 31;
    int ntiles = (N + 15) >> 4;
    if (wid >= ntiles) return;

    int m    = lane & 15;
    int hf   = lane >> 4;
    int nlo  = lane & 15;
    int rowg = wid * 16 + m;
    bool full = (wid * 16 + 16 <= N);     // single per-wave guard
    int rowc = full ? rowg : (rowg < N ? rowg : N - 1);

    // A: two 16x32 bf16 K-chunks; per chunk two runs of 8 consecutive k
    v16bf a0, a1;
    {
        const float* src = x_last + (long long)rowc * D_H + hf * 8;
        const float4* p  = (const float4*)src;
        put8(a0, 0, p[0], p[1]);          // k = hf*8 .. +7
        put8(a0, 8, p[4], p[5]);          // k = hf*8+16 .. +23
        put8(a1, 0, p[8], p[9]);          // +32
        put8(a1, 8, p[12], p[13]);
    }
    if (!full && rowg >= N) {
        #pragma unroll
        for (int i = 0; i < 16; ++i) { a0[i] = (__bf16)0.0f; a1[i] = (__bf16)0.0f; }
    }

    v8f c0 = {}, c1 = {}, c2 = {}, c3 = {};
    #pragma unroll
    for (int nt = 0; nt < 4; ++nt) {
        const v16bf b0 = *reinterpret_cast<const v16bf*>(&frag[((0 * 4 + nt) * 32 + lane) * 8]);
        const v16bf b1 = *reinterpret_cast<const v16bf*>(&frag[((1 * 4 + nt) * 32 + lane) * 8]);
        v8f acc = (nt == 0) ? c0 : (nt == 1) ? c1 : (nt == 2) ? c2 : c3;
        acc = __builtin_amdgcn_wmma_f32_16x16x32_bf16(false, a0, false, b0, (short)0, acc, false, false);
        acc = __builtin_amdgcn_wmma_f32_16x16x32_bf16(false, a1, false, b1, (short)0, acc, false, false);
        if (nt == 0) c0 = acc; else if (nt == 1) c1 = acc; else if (nt == 2) c2 = acc; else c3 = acc;
    }

    float* outbase = xw + (long long)(wid * 16 + hf * 8) * D_H + nlo;
    if (full) {
        #pragma unroll
        for (int nt = 0; nt < 4; ++nt) {
            v8f acc = (nt == 0) ? c0 : (nt == 1) ? c1 : (nt == 2) ? c2 : c3;
            #pragma unroll
            for (int r = 0; r < 8; ++r)
                outbase[(long long)r * D_H + nt * 16] = acc[r];
        }
    } else {
        #pragma unroll
        for (int nt = 0; nt < 4; ++nt) {
            v8f acc = (nt == 0) ? c0 : (nt == 1) ? c1 : (nt == 2) ? c2 : c3;
            #pragma unroll
            for (int r = 0; r < 8; ++r) {
                int mrow = wid * 16 + hf * 8 + r;
                if (mrow < N) xw[(long long)mrow * D_H + nt * 16 + nlo] = acc[r];
            }
        }
    }
}

// ---------------------------------------------------------------------------
// Scatter: h[col] += dis[row]*dis[col] * xw[row]   (edges + self loops)
// ---------------------------------------------------------------------------
__global__ void scatter_msg(const float* __restrict__ xw,
                            const float* __restrict__ dis,
                            const int* __restrict__ row,
                            const int* __restrict__ col,
                            float* __restrict__ h,
                            int E, int N) {
    long long idx = (long long)blockIdx.x * blockDim.x + threadIdx.x;
    long long tot = (long long)(E + N) * 16;
    if (idx >= tot) return;
    long long e = idx >> 4;
    int fo = (int)(idx & 15) * 4;
    int r, c;
    if (e < E) { r = row[e]; c = col[e]; }
    else       { r = c = (int)(e - E); }
    float nrm = dis[r] * dis[c];
    const float4 v = *(const float4*)&xw[(long long)r * D_H + fo];
    float* dst = &h[(long long)c * D_H + fo];
    atomicAdd(dst + 0, nrm * v.x);
    atomicAdd(dst + 1, nrm * v.y);
    atomicAdd(dst + 2, nrm * v.z);
    atomicAdd(dst + 3, nrm * v.w);
}

// ---------------------------------------------------------------------------
// Edge MLP: out[e] = relu(concat(relu h[row], relu h[col], ea) @ W1 + b1) @ w2 + b2
// K padded 144->160, bias folded at k==144 (feature == 1). Wave per 16 edges.
// W1 staged in LDS pre-swizzled into WMMA B-fragment layout (20 KB).
// ---------------------------------------------------------------------------
__global__ void edge_mlp_wmma(const float* __restrict__ h,
                              const float* __restrict__ ea,
                              const int* __restrict__ row,
                              const int* __restrict__ col,
                              const float* __restrict__ w1,
                              const float* __restrict__ b1,
                              const float* __restrict__ w2,
                              const float* __restrict__ b2,
                              float* __restrict__ out, int E) {
    __shared__ unsigned frag[5 * 4 * 32 * 8];   // 20 KB
    __shared__ float    W2l[D_H];
    for (int idx = threadIdx.x; idx < 5 * 4 * 32 * 8; idx += blockDim.x) {
        int j    = idx & 7;
        int lane = (idx >> 3) & 31;
        int nt   = (idx >> 8) & 3;
        int kc   = idx >> 10;
        int n  = nt * 16 + (lane & 15);
        int k0 = kc * 32 + (lane >> 4) * 8 + (j >> 2) * 16 + (j & 3) * 2;
        int k1 = k0 + 1;
        float v0 = (k0 < 144) ? w1[k0 * D_H + n] : (k0 == 144 ? b1[n] : 0.0f);
        float v1 = (k1 < 144) ? w1[k1 * D_H + n] : (k1 == 144 ? b1[n] : 0.0f);
        frag[idx] = pack2bf(v0, v1);
    }
    if (threadIdx.x < D_H) W2l[threadIdx.x] = w2[threadIdx.x];
    __syncthreads();

    int wid  = (blockIdx.x * blockDim.x + threadIdx.x) >> 5;
    int lane = threadIdx.x & 31;
    int etiles = (E + 15) >> 4;
    if (wid >= etiles) return;

    int m   = lane & 15;
    int hf  = lane >> 4;
    int nlo = lane & 15;
    long long e = (long long)wid * 16 + m;
    if (e >= E) e = E - 1;
    int ri = row[e], ci = col[e];

    const float* hr = h + (long long)ri * D_H;
    const float* hc = h + (long long)ci * D_H;
    const float* ee = ea + e * D_E;

    v8f c0 = {}, c1 = {}, c2 = {}, c3 = {};
    #pragma unroll
    for (int kc = 0; kc < 5; ++kc) {
        v16bf a;
        if (kc < 4) {
            const float* src = ((kc < 2) ? hr + kc * 32 : hc + (kc - 2) * 32) + hf * 8;
            const float4* p  = (const float4*)src;
            put8(a, 0, relu4(p[0]), relu4(p[1]));   // k = base .. +7
            put8(a, 8, relu4(p[4]), relu4(p[5]));   // k = base+16 .. +23
        } else {
            const float4* p = (const float4*)(ee + hf * 8);
            put8(a, 0, p[0], p[1]);                 // edge_attr (k 128..143)
            float4 one  = {1.0f, 0.0f, 0.0f, 0.0f};
            float4 zero = {0.0f, 0.0f, 0.0f, 0.0f};
            if (hf == 0) put8(a, 8, one, zero);     // k 144 -> bias feature
            else         put8(a, 8, zero, zero);    // k 152..159 padding
        }
        #pragma unroll
        for (int nt = 0; nt < 4; ++nt) {
            const v16bf b = *reinterpret_cast<const v16bf*>(&frag[((kc * 4 + nt) * 32 + lane) * 8]);
            v8f acc = (nt == 0) ? c0 : (nt == 1) ? c1 : (nt == 2) ? c2 : c3;
            acc = __builtin_amdgcn_wmma_f32_16x16x32_bf16(false, a, false, b, (short)0, acc, false, false);
            if (nt == 0) c0 = acc; else if (nt == 1) c1 = acc; else if (nt == 2) c2 = acc; else c3 = acc;
        }
    }

    // out[e] = sum_n relu(hidden[m][n]) * w2[n] + b2
    float s[8];
    #pragma unroll
    for (int r = 0; r < 8; ++r) s[r] = 0.0f;
    #pragma unroll
    for (int nt = 0; nt < 4; ++nt) {
        v8f acc = (nt == 0) ? c0 : (nt == 1) ? c1 : (nt == 2) ? c2 : c3;
        float w2v = W2l[nt * 16 + nlo];
        #pragma unroll
        for (int r = 0; r < 8; ++r) s[r] += fmaxf(acc[r], 0.0f) * w2v;
    }
    #pragma unroll
    for (int r = 0; r < 8; ++r) {
        #pragma unroll
        for (int mm = 8; mm >= 1; mm >>= 1) s[r] += __shfl_xor(s[r], mm, 16);
    }
    if (nlo == 0) {
        float bb = b2[0];
        long long base = (long long)wid * 16 + hf * 8;
        if (base + 8 <= E) {
            float4 o0 = {s[0] + bb, s[1] + bb, s[2] + bb, s[3] + bb};
            float4 o1 = {s[4] + bb, s[5] + bb, s[6] + bb, s[7] + bb};
            *(float4*)&out[base]     = o0;
            *(float4*)&out[base + 4] = o1;
        } else {
            #pragma unroll
            for (int r = 0; r < 8; ++r)
                if (base + r < E) out[base + r] = s[r] + bb;
        }
    }
}

// ---------------------------------------------------------------------------
extern "C" void kernel_launch(void* const* d_in, const int* in_sizes, int n_in,
                              void* d_out, int out_size, void* d_ws, size_t ws_size,
                              hipStream_t stream) {
    const float* x_seq  = (const float*)d_in[0];
    const float* ea     = (const float*)d_in[1];
    const float* init_w = (const float*)d_in[2];
    const float* W1     = (const float*)d_in[3];
    const float* B1     = (const float*)d_in[4];
    const float* W2     = (const float*)d_in[5];
    const float* B2     = (const float*)d_in[6];
    const float* mw1    = (const float*)d_in[7];
    const float* mb1    = (const float*)d_in[8];
    const float* mw2    = (const float*)d_in[9];
    const float* mb2    = (const float*)d_in[10];
    const int*   eidx   = (const int*)d_in[11];

    const int E = in_sizes[1] / D_E;
    const int N = in_sizes[0] / (T_STEPS * D_H);
    const int* row = eidx;
    const int* col = eidx + E;
    const float* x_last = x_seq + (long long)(T_STEPS - 1) * N * D_H;
    float* out = (float*)d_out;

    float* ws    = (float*)d_ws;
    float* w_cur = ws;
    float* k1    = ws + 1 * FLAT;
    float* k2    = ws + 2 * FLAT;
    float* k3    = ws + 3 * FLAT;
    float* k4    = ws + 4 * FLAT;
    float* argv  = ws + 5 * FLAT;
    float* hid   = ws + 6 * FLAT;
    float* deg   = ws + 6 * FLAT + BNECK;
    float* xw    = deg + N;
    float* hbuf  = xw + (long long)N * D_H;

    const int TB = 256;

    // init + degree
    {
        long long tot = (long long)N * D_H;
        init_state<<<(int)((tot + TB - 1) / TB), TB, 0, stream>>>(init_w, w_cur, deg, hbuf, N);
        deg_accum<<<(E + TB - 1) / TB, TB, 0, stream>>>(col, deg, E);
        deg_to_dis<<<(N + TB - 1) / TB, TB, 0, stream>>>(deg, N);
    }

    // weight ODE: 7 RK4 (3/8-rule alt) steps
    const float hh = 1.0f / (float)(T_STEPS - 1);
    const int argB = (FLAT + TB - 1) / TB;
    const int hidB = (BNECK * 32 + TB - 1) / TB;
    const int outB = (FLAT * 32 + TB - 1) / TB;
    for (int s = 0; s < T_STEPS - 1; ++s) {
        ode_arg<<<argB, TB, 0, stream>>>(w_cur, k1, k1, k1, 0.0f, 0.0f, 0.0f, argv);
        ode_hidden<<<hidB, TB, 0, stream>>>(argv, W1, B1, hid);
        ode_out<<<outB, TB, 0, stream>>>(hid, W2, B2, k1);

        ode_arg<<<argB, TB, 0, stream>>>(w_cur, k1, k1, k1, hh / 3.0f, 0.0f, 0.0f, argv);
        ode_hidden<<<hidB, TB, 0, stream>>>(argv, W1, B1, hid);
        ode_out<<<outB, TB, 0, stream>>>(hid, W2, B2, k2);

        ode_arg<<<argB, TB, 0, stream>>>(w_cur, k1, k2, k2, -hh / 3.0f, hh, 0.0f, argv);
        ode_hidden<<<hidB, TB, 0, stream>>>(argv, W1, B1, hid);
        ode_out<<<outB, TB, 0, stream>>>(hid, W2, B2, k3);

        ode_arg<<<argB, TB, 0, stream>>>(w_cur, k1, k2, k3, hh, -hh, hh, argv);
        ode_hidden<<<hidB, TB, 0, stream>>>(argv, W1, B1, hid);
        ode_out<<<outB, TB, 0, stream>>>(hid, W2, B2, k4);

        ode_combine<<<argB, TB, 0, stream>>>(w_cur, k1, k2, k3, k4, hh);
    }

    // xw = x_last @ W_T  (WMMA)
    {
        int ntiles = (N + 15) >> 4;
        int blocks = (ntiles * 32 + TB - 1) / TB;
        xw_wmma<<<blocks, TB, 0, stream>>>(x_last, w_cur, xw, N);
    }

    // scatter-add messages (incl. self loops)
    {
        long long tot = (long long)(E + N) * 16;
        scatter_msg<<<(int)((tot + TB - 1) / TB), TB, 0, stream>>>(xw, deg, row, col, hbuf, E, N);
    }

    // edge MLP (WMMA), relu on h fused into the gather
    {
        int etiles = (E + 15) >> 4;
        int blocks = (etiles * 32 + TB - 1) / TB;
        edge_mlp_wmma<<<blocks, TB, 0, stream>>>(hbuf, ea, row, col, mw1, mb1, mw2, mb2, out, E);
    }
}
